// tfdiff_WiFi_LoRA_9990093930835
// MI455X (gfx1250) — matrix operand order repaired and checked by reading, not compiled, not run on hardware
//
#include <hip/hip_runtime.h>

// ---------------------------------------------------------------------------
// Complex DiT forward (tfdiff WiFi LoRA) for gfx1250.
// All GEMMs run through v_wmma_f32_16x16x32_f16 (f16 in, f32 accumulate).
// LoRA adapters are merged into the QKV weights on-device each layer.
// v2: double-buffered LDS, guard-free vectorized staging for interior tiles,
//     coalesced transposed-B staging, global_prefetch of tile k+2.
// ---------------------------------------------------------------------------

typedef __attribute__((ext_vector_type(16))) _Float16 v16h;
typedef __attribute__((ext_vector_type(8)))  _Float16 v8h;
typedef __attribute__((ext_vector_type(2)))  _Float16 v2h;
typedef __attribute__((ext_vector_type(8)))  float    v8f;

#define TF_B    16
#define TF_N    512
#define TF_H    512
#define TF_NH   8
#define TF_L    8
#define TF_R    8
#define TF_IN   90
#define TF_COND 128
#define TF_EMB  256
#define TF_MLP  (4 * TF_H)
#define TF_DH   (TF_H / TF_NH)     // 64
#define TF_M    (TF_B * TF_N)      // 8192
#define TF_LORA_SCALE (16.0f / TF_R)

#define F_SILU 1
#define F_BIAS 2
#define F_POS  4
#define F_GATE 8

struct GemmP {
  const float* A;  const float* Bm; const float* bias; float* Y;
  long long ldA, ldB, ldY;                 // leading dims in complex elements
  long long aSO, aSI, bSO, bSI, ySO, ySI;  // per-z offsets (complex elements)
  int aDiv, bDiv, yDiv;                    // z -> (z/div)*SO + (z%div)*SI
  int bTrans;                              // 1: B stored [K][N] instead of [N][K]
  int M, N, K, flags;
  float scale;
  const float* pos; int seqN;              // F_POS: += pos[(row%seqN)*N + col]
  const float* gate; long long gateStride; // F_GATE: y = Y + gate[row/seqN] * y
};

// ---------------------------------------------------------------------------
// Generic batched complex GEMM:  Y[m,n] = sum_k A[m,k] * W[n,k]  (complex),
// tiled 64x64x32, 8 waves/block, each wave owns one 16-row group and two
// 16-col groups -> 8 v_wmma per K-step.  LDS is ping-pong double buffered.
// ---------------------------------------------------------------------------
__launch_bounds__(256)
__global__ void cgemm_wmma(GemmP p) {
  __shared__ _Float16 As[2][2][64 * 32];   // [buf][r/i][m][k]
  __shared__ _Float16 Bs[2][2][64 * 32];   // [buf][r/i][n][k]   (n-major!)

  const int z = blockIdx.z;
  const float* Ab = p.A  + (((long long)(z / p.aDiv) * p.aSO + (long long)(z % p.aDiv) * p.aSI) << 1);
  const float* Bb = p.Bm + (((long long)(z / p.bDiv) * p.bSO + (long long)(z % p.bDiv) * p.bSI) << 1);
  float*       Yb = p.Y  + (((long long)(z / p.yDiv) * p.ySO + (long long)(z % p.yDiv) * p.ySI) << 1);

  const int rowBase = blockIdx.y << 6;
  const int colBase = blockIdx.x << 6;
  const int tid  = threadIdx.x;
  const int wave = tid >> 5;
  const int lane = tid & 31;
  const int lh   = lane >> 4;     // half-wave select
  const int ln   = lane & 15;
  const int mi   = wave & 3;           // A 16-row group (0..3)
  const int nj0  = (wave >> 2) << 1;   // first of two B 16-col groups

  // block-uniform predicate: full 64x64 tile in M/N (K checked per k-step)
  const bool fullMN = (rowBase + 64 <= p.M) && (colBase + 64 <= p.N);

  // --- tile staging: global f32 complex -> two f16 LDS planes ---------------
  auto stage = [&](int buf, int k0) {
    const bool fast = fullMN && (k0 + 32 <= p.K);
    if (fast) {
      // A: 2 complex (float4, 16B, coalesced) per thread x4
#pragma unroll
      for (int i = 0; i < 4; ++i) {
        const int m = (tid >> 4) + 16 * i;
        const int k = (tid & 15) * 2;
        const float4 v4 = *(const float4*)(Ab + (((long long)(rowBase + m) * p.ldA + k0 + k) << 1));
        v2h hr = {(_Float16)v4.x, (_Float16)v4.z};
        v2h hi = {(_Float16)v4.y, (_Float16)v4.w};
        *(v2h*)&As[buf][0][m * 32 + k] = hr;
        *(v2h*)&As[buf][1][m * 32 + k] = hi;
      }
      if (!p.bTrans) {
#pragma unroll
        for (int i = 0; i < 4; ++i) {
          const int n = (tid >> 4) + 16 * i;
          const int k = (tid & 15) * 2;
          const float4 v4 = *(const float4*)(Bb + (((long long)(colBase + n) * p.ldB + k0 + k) << 1));
          v2h hr = {(_Float16)v4.x, (_Float16)v4.z};
          v2h hi = {(_Float16)v4.y, (_Float16)v4.w};
          *(v2h*)&Bs[buf][0][n * 32 + k] = hr;
          *(v2h*)&Bs[buf][1][n * 32 + k] = hi;
        }
      } else {
        // B stored [K][N]: contiguous along n -> pair over n
#pragma unroll
        for (int i = 0; i < 4; ++i) {
          const int e = tid + (i << 8);         // 0..1023 pairs
          const int k = e >> 5;                 // 0..31
          const int n = (e & 31) * 2;           // 0..62 even
          const float4 v4 = *(const float4*)(Bb + (((long long)(k0 + k) * p.ldB + colBase + n) << 1));
          Bs[buf][0][n * 32 + k]       = (_Float16)v4.x;
          Bs[buf][1][n * 32 + k]       = (_Float16)v4.y;
          Bs[buf][0][(n + 1) * 32 + k] = (_Float16)v4.z;
          Bs[buf][1][(n + 1) * 32 + k] = (_Float16)v4.w;
        }
      }
    } else {
      // guarded scalar path (edge tiles only)
#pragma unroll
      for (int i = 0; i < 8; ++i) {
        const int e = tid + (i << 8);
        const int m = e >> 5, k = e & 31;
        const int gm = rowBase + m, gk = k0 + k;
        float vr = 0.f, vi = 0.f;
        if (gm < p.M && gk < p.K) {
          const float* s = Ab + (((long long)gm * p.ldA + gk) << 1);
          vr = s[0]; vi = s[1];
        }
        As[buf][0][m * 32 + k] = (_Float16)vr;
        As[buf][1][m * 32 + k] = (_Float16)vi;
      }
#pragma unroll
      for (int i = 0; i < 8; ++i) {
        const int e = tid + (i << 8);
        int n, k;
        if (p.bTrans) { k = e >> 6; n = e & 63; }      // coalesced along n
        else          { n = e >> 5; k = e & 31; }
        const int gn = colBase + n, gk = k0 + k;
        float vr = 0.f, vi = 0.f;
        if (gn < p.N && gk < p.K) {
          const float* s = p.bTrans ? (Bb + (((long long)gk * p.ldB + gn) << 1))
                                    : (Bb + (((long long)gn * p.ldB + gk) << 1));
          vr = s[0]; vi = s[1];
        }
        Bs[buf][0][n * 32 + k] = (_Float16)vr;
        Bs[buf][1][n * 32 + k] = (_Float16)vi;
      }
    }
  };

  const v8f zf = {0.f, 0.f, 0.f, 0.f, 0.f, 0.f, 0.f, 0.f};
  v8f acc_rr[2] = {zf, zf}, acc_ii[2] = {zf, zf};
  v8f acc_ri[2] = {zf, zf}, acc_ir[2] = {zf, zf};

  const int kTiles = (p.K + 31) >> 5;
  stage(0, 0);
  __syncthreads();

  for (int kt = 0; kt < kTiles; ++kt) {
    const int cur = kt & 1;
    if (kt + 1 < kTiles) stage(cur ^ 1, (kt + 1) << 5);
    if (kt + 2 < kTiles) {
      // warm GL2 for the tile after next (lowers to global_prefetch_b8)
      const int kp = (kt + 2) << 5;
      __builtin_prefetch(Ab + (((long long)(rowBase + (tid >> 4)) * p.ldA + kp) << 1), 0, 1);
      if (p.bTrans)
        __builtin_prefetch(Bb + (((long long)(kp + (tid >> 4)) * p.ldB + colBase) << 1), 0, 1);
      else
        __builtin_prefetch(Bb + (((long long)(colBase + (tid >> 4)) * p.ldB + kp) << 1), 0, 1);
    }

    // A fragment: lane row = mi*16+ln ; chunks at k = 8*lh and 16+8*lh
    const _Float16* apr = &As[cur][0][(mi * 16 + ln) * 32];
    const _Float16* api = &As[cur][1][(mi * 16 + ln) * 32];
    v8h arl = *(const v8h*)(apr + 8 * lh);
    v8h arh = *(const v8h*)(apr + 16 + 8 * lh);
    v8h ail = *(const v8h*)(api + 8 * lh);
    v8h aih = *(const v8h*)(api + 16 + 8 * lh);
    v16h Ar = __builtin_shufflevector(arl, arh, 0,1,2,3,4,5,6,7,8,9,10,11,12,13,14,15);
    v16h Ai = __builtin_shufflevector(ail, aih, 0,1,2,3,4,5,6,7,8,9,10,11,12,13,14,15);

#pragma unroll
    for (int j = 0; j < 2; ++j) {
      // B fragment: lane col = group*16+ln ; chunks at k = 8*lh and 16+8*lh
      const _Float16* bpr = &Bs[cur][0][((nj0 + j) * 16 + ln) * 32] + 8 * lh;
      const _Float16* bpi = &Bs[cur][1][((nj0 + j) * 16 + ln) * 32] + 8 * lh;
      v8h brl = *(const v8h*)bpr, brh = *(const v8h*)(bpr + 16);
      v8h bil = *(const v8h*)bpi, bih = *(const v8h*)(bpi + 16);
      v16h Br = __builtin_shufflevector(brl, brh, 0,1,2,3,4,5,6,7,8,9,10,11,12,13,14,15);
      v16h Bi = __builtin_shufflevector(bil, bih, 0,1,2,3,4,5,6,7,8,9,10,11,12,13,14,15);
      acc_rr[j] = __builtin_amdgcn_wmma_f32_16x16x32_f16(false, Ar, false, Br, (short)0, acc_rr[j], false, false);
      acc_ii[j] = __builtin_amdgcn_wmma_f32_16x16x32_f16(false, Ai, false, Bi, (short)0, acc_ii[j], false, false);
      acc_ri[j] = __builtin_amdgcn_wmma_f32_16x16x32_f16(false, Ar, false, Bi, (short)0, acc_ri[j], false, false);
      acc_ir[j] = __builtin_amdgcn_wmma_f32_16x16x32_f16(false, Ai, false, Br, (short)0, acc_ir[j], false, false);
    }
    __syncthreads();
  }

  // ---- epilogue: combine complex parts, bias / silu / pos / gated residual ----
#pragma unroll
  for (int j = 0; j < 2; ++j) {
    const int gcol = colBase + (nj0 + j) * 16 + ln;
    if (gcol >= p.N) continue;
    float bre = 0.f, bim = 0.f;
    if (p.flags & F_BIAS) { bre = p.bias[2 * gcol]; bim = p.bias[2 * gcol + 1]; }
#pragma unroll
    for (int v = 0; v < 8; ++v) {
      const int grow = rowBase + mi * 16 + 8 * lh + v;   // C layout: m = v + 8*lh
      if (grow >= p.M) continue;
      float yr = (acc_rr[j][v] - acc_ii[j][v]) * p.scale + bre;
      float yi = (acc_ri[j][v] + acc_ir[j][v]) * p.scale + bim;
      if (p.flags & F_SILU) {
        yr = yr / (1.f + __expf(-yr));
        yi = yi / (1.f + __expf(-yi));
      }
      if (p.flags & F_POS) {
        const float* pp = p.pos + (((long long)(grow % p.seqN) * p.N + gcol) << 1);
        yr += pp[0]; yi += pp[1];
      }
      float* dst = Yb + (((long long)grow * p.ldY + gcol) << 1);
      if (p.flags & F_GATE) {   // elementwise (per r/i channel) gate + residual
        const int b = grow / p.seqN;
        const float* g = p.gate + (long long)b * p.gateStride + 2 * gcol;
        yr = dst[0] + g[0] * yr;
        yi = dst[1] + g[1] * yi;
      }
      dst[0] = yr; dst[1] = yi;
    }
  }
}

// ---------------------------------------------------------------------------
// LoRA weight merge: out[o,i] = w[o,i] + s * sum_r bw[o,r]*aw[r,i]  (complex)
// ---------------------------------------------------------------------------
__global__ void merge_lora_kernel(const float* __restrict__ w,
                                  const float* __restrict__ bw,
                                  const float* __restrict__ aw,
                                  float* __restrict__ out, float s) {
  const int idx = blockIdx.x * 256 + threadIdx.x;     // o*H + i
  if (idx >= TF_H * TF_H) return;
  const int o = idx >> 9, ii = idx & 511;
  float accR = 0.f, accI = 0.f;
#pragma unroll
  for (int r = 0; r < TF_R; ++r) {
    const float br = bw[(o * TF_R + r) * 2], bi = bw[(o * TF_R + r) * 2 + 1];
    const float ar = aw[(r * TF_H + ii) * 2], ai = aw[(r * TF_H + ii) * 2 + 1];
    accR += br * ar - bi * ai;
    accI += br * ai + bi * ar;
  }
  out[2 * idx]     = w[2 * idx]     + s * accR;
  out[2 * idx + 1] = w[2 * idx + 1] + s * accI;
}

// ---------------------------------------------------------------------------
// Complex LayerNorm (r/i channels normalized independently over H) + modulate.
// One block per token.
// ---------------------------------------------------------------------------
__launch_bounds__(256)
__global__ void cln_mod_kernel(const float* __restrict__ X, float* __restrict__ Y,
                               const float* __restrict__ shiftB,
                               const float* __restrict__ scaleB,
                               int condStride, int seqN) {
  __shared__ float4 red[256];
  const int token = blockIdx.x;
  const int b = token / seqN;
  const float* row = X + (size_t)token * TF_H * 2;
  float sr = 0.f, si = 0.f, qr = 0.f, qi = 0.f;
  for (int f = threadIdx.x; f < TF_H; f += 256) {
    const float xr = row[2 * f], xi = row[2 * f + 1];
    sr += xr; si += xi; qr += xr * xr; qi += xi * xi;
  }
  red[threadIdx.x] = make_float4(sr, si, qr, qi);
  __syncthreads();
  for (int st = 128; st > 0; st >>= 1) {
    if (threadIdx.x < st) {
      float4 a = red[threadIdx.x], c = red[threadIdx.x + st];
      red[threadIdx.x] = make_float4(a.x + c.x, a.y + c.y, a.z + c.z, a.w + c.w);
    }
    __syncthreads();
  }
  const float4 tot = red[0];
  const float mur = tot.x / TF_H, mui = tot.y / TF_H;
  const float vr = tot.z / TF_H - mur * mur;
  const float vi = tot.w / TF_H - mui * mui;
  const float ir = rsqrtf(vr + 1e-6f), ii2 = rsqrtf(vi + 1e-6f);
  const float* sh = shiftB + (size_t)b * condStride;
  const float* sc = scaleB + (size_t)b * condStride;
  float* out = Y + (size_t)token * TF_H * 2;
  for (int f = threadIdx.x; f < TF_H; f += 256) {
    const float nr = (row[2 * f]     - mur) * ir;
    const float ni = (row[2 * f + 1] - mui) * ii2;
    out[2 * f]     = nr * (1.f + sc[2 * f])     + sh[2 * f];
    out[2 * f + 1] = ni * (1.f + sc[2 * f + 1]) + sh[2 * f + 1];
  }
}

// ---------------------------------------------------------------------------
// Softmax over keys, r/i channels independent; in-place on [.., Nk, 2] rows.
// One block per (batch-head, query) row.
// ---------------------------------------------------------------------------
__launch_bounds__(256)
__global__ void softmax2_kernel(float* __restrict__ S) {
  const int Nk = TF_N;
  float* row = S + (size_t)blockIdx.x * Nk * 2;
  __shared__ float2 red[256];
  float mr = -3.4e38f, mi = -3.4e38f;
  for (int k = threadIdx.x; k < Nk; k += 256) {
    mr = fmaxf(mr, row[2 * k]);
    mi = fmaxf(mi, row[2 * k + 1]);
  }
  red[threadIdx.x] = make_float2(mr, mi);
  __syncthreads();
  for (int st = 128; st > 0; st >>= 1) {
    if (threadIdx.x < st) {
      float2 a = red[threadIdx.x], c = red[threadIdx.x + st];
      red[threadIdx.x] = make_float2(fmaxf(a.x, c.x), fmaxf(a.y, c.y));
    }
    __syncthreads();
  }
  const float2 mx = red[0];
  __syncthreads();
  float sr = 0.f, si = 0.f;
  for (int k = threadIdx.x; k < Nk; k += 256) {
    const float er = __expf(row[2 * k] - mx.x);
    const float ei = __expf(row[2 * k + 1] - mx.y);
    row[2 * k] = er; row[2 * k + 1] = ei;
    sr += er; si += ei;
  }
  red[threadIdx.x] = make_float2(sr, si);
  __syncthreads();
  for (int st = 128; st > 0; st >>= 1) {
    if (threadIdx.x < st) {
      float2 a = red[threadIdx.x], c = red[threadIdx.x + st];
      red[threadIdx.x] = make_float2(a.x + c.x, a.y + c.y);
    }
    __syncthreads();
  }
  const float invr = 1.f / red[0].x, invi = 1.f / red[0].y;
  for (int k = threadIdx.x; k < Nk; k += 256) {
    row[2 * k] *= invr; row[2 * k + 1] *= invi;
  }
}

// ---------------------------------------------------------------------------
__global__ void gather_t_kernel(const int* __restrict__ t,
                                const float* __restrict__ ttab,
                                float* __restrict__ out) {
  const int idx = blockIdx.x * 256 + threadIdx.x;   // B * EMB * 2 floats
  if (idx >= TF_B * TF_EMB * 2) return;
  const int b = idx / (TF_EMB * 2);
  const int r = idx % (TF_EMB * 2);
  out[idx] = ttab[(size_t)t[b] * (TF_EMB * 2) + r];
}

__global__ void add_silu_kernel(const float* __restrict__ a,
                                const float* __restrict__ b,
                                float* __restrict__ out, int n) {
  const int i = blockIdx.x * 256 + threadIdx.x;
  if (i < n) {
    const float v = a[i] + b[i];
    out[i] = v / (1.f + __expf(-v));
  }
}

// ---------------------------------------------------------------------------
static GemmP mk(const float* A, long long ldA, const float* Bm, long long ldB,
                const float* bias, float* Y, long long ldY,
                int M, int N, int K, int flags, float scale) {
  GemmP p;
  p.A = A; p.Bm = Bm; p.bias = bias; p.Y = Y;
  p.ldA = ldA; p.ldB = ldB; p.ldY = ldY;
  p.aSO = 0; p.aSI = 0; p.bSO = 0; p.bSI = 0; p.ySO = 0; p.ySI = 0;
  p.aDiv = 1; p.bDiv = 1; p.yDiv = 1; p.bTrans = 0;
  p.M = M; p.N = N; p.K = K; p.flags = flags; p.scale = scale;
  p.pos = nullptr; p.seqN = TF_N; p.gate = nullptr; p.gateStride = 0;
  return p;
}

static void run_gemm(hipStream_t s, const GemmP& p, int Z) {
  dim3 g((unsigned)((p.N + 63) / 64), (unsigned)((p.M + 63) / 64), (unsigned)Z);
  cgemm_wmma<<<g, dim3(256), 0, s>>>(p);
}

extern "C" void kernel_launch(void* const* d_in, const int* in_sizes, int n_in,
                              void* d_out, int out_size, void* d_ws, size_t ws_size,
                              hipStream_t stream) {
  (void)in_sizes; (void)n_in; (void)out_size;

  const float* x     = (const float*)d_in[0];
  const int*   tIdx  = (const int*)  d_in[1];
  const float* c     = (const float*)d_in[2];
  const float* p_w   = (const float*)d_in[3];
  const float* p_b   = (const float*)d_in[4];
  const float* pos   = (const float*)d_in[5];
  const float* ttab  = (const float*)d_in[6];
  const float* t_w1  = (const float*)d_in[7];
  const float* t_b1  = (const float*)d_in[8];
  const float* t_w2  = (const float*)d_in[9];
  const float* t_b2  = (const float*)d_in[10];
  const float* c_w1  = (const float*)d_in[11];
  const float* c_b1  = (const float*)d_in[12];
  const float* c_w2  = (const float*)d_in[13];
  const float* c_b2  = (const float*)d_in[14];
  const float* ada_w = (const float*)d_in[15];
  const float* ada_b = (const float*)d_in[16];
  const float* wq_w  = (const float*)d_in[17];
  const float* wq_b  = (const float*)d_in[18];
  const float* wk_w  = (const float*)d_in[19];
  const float* wk_b  = (const float*)d_in[20];
  const float* wv_w  = (const float*)d_in[21];
  const float* wv_b  = (const float*)d_in[22];
  const float* qa    = (const float*)d_in[23];
  const float* qb    = (const float*)d_in[24];
  const float* ka    = (const float*)d_in[25];
  const float* kb    = (const float*)d_in[26];
  const float* va    = (const float*)d_in[27];
  const float* vb    = (const float*)d_in[28];
  const float* wo_w  = (const float*)d_in[29];
  const float* wo_b  = (const float*)d_in[30];
  const float* m_w1  = (const float*)d_in[31];
  const float* m_b1  = (const float*)d_in[32];
  const float* m_w2  = (const float*)d_in[33];
  const float* m_b2  = (const float*)d_in[34];
  const float* fadaw = (const float*)d_in[35];
  const float* fadab = (const float*)d_in[36];
  const float* f_w   = (const float*)d_in[37];
  const float* f_b   = (const float*)d_in[38];

  // -------- workspace carve (floats) --------
  float* w = (float*)d_ws;
  auto alloc = [&](size_t n) { float* p = w; w += n; return p; };
  float* hBuf   = alloc((size_t)TF_M * TF_H * 2);
  float* hmBuf  = alloc((size_t)TF_M * TF_H * 2);
  float* qBuf   = alloc((size_t)TF_M * TF_H * 2);
  float* kBuf   = alloc((size_t)TF_M * TF_H * 2);
  float* vBuf   = alloc((size_t)TF_M * TF_H * 2);
  float* aoBuf  = alloc((size_t)TF_M * TF_H * 2);
  float* mlpBuf = alloc((size_t)TF_M * TF_MLP * 2);
  float* scoreB = alloc((size_t)TF_B * TF_NH * TF_N * TF_N * 2);
  float* wqe    = alloc((size_t)TF_H * TF_H * 2);
  float* wke    = alloc((size_t)TF_H * TF_H * 2);
  float* wve    = alloc((size_t)TF_H * TF_H * 2);
  float* tEmb   = alloc((size_t)TF_B * TF_EMB * 2);
  float* tmp1   = alloc((size_t)TF_B * TF_H * 2);
  float* teBuf  = alloc((size_t)TF_B * TF_H * 2);
  float* ceBuf  = alloc((size_t)TF_B * TF_H * 2);
  float* scond  = alloc((size_t)TF_B * TF_H * 2);   // silu(cond)
  float* modBuf = alloc((size_t)TF_B * 6 * TF_H * 2);
  float* fmodB  = alloc((size_t)TF_B * 2 * TF_H * 2);
  if ((size_t)((char*)w - (char*)d_ws) > ws_size) return;  // ws too small

  // -------- embeddings --------
  {
    GemmP pe = mk(x, TF_IN, p_w, TF_IN, p_b, hBuf, TF_H,
                  TF_M, TF_H, TF_IN, F_BIAS | F_POS, 1.f);
    pe.pos = pos; pe.seqN = TF_N;
    run_gemm(stream, pe, 1);
  }
  gather_t_kernel<<<(TF_B * TF_EMB * 2 + 255) / 256, 256, 0, stream>>>(tIdx, ttab, tEmb);
  run_gemm(stream, mk(tEmb, TF_EMB, t_w1, TF_EMB, t_b1, tmp1, TF_H,
                      TF_B, TF_H, TF_EMB, F_BIAS | F_SILU, 1.f), 1);
  run_gemm(stream, mk(tmp1, TF_H, t_w2, TF_H, t_b2, teBuf, TF_H,
                      TF_B, TF_H, TF_H, F_BIAS | F_SILU, 1.f), 1);
  run_gemm(stream, mk(c, TF_COND, c_w1, TF_COND, c_b1, tmp1, TF_H,
                      TF_B, TF_H, TF_COND, F_BIAS | F_SILU, 1.f), 1);
  run_gemm(stream, mk(tmp1, TF_H, c_w2, TF_H, c_b2, ceBuf, TF_H,
                      TF_B, TF_H, TF_H, F_BIAS, 1.f), 1);
  add_silu_kernel<<<(TF_B * TF_H * 2 + 255) / 256, 256, 0, stream>>>(
      ceBuf, teBuf, scond, TF_B * TF_H * 2);

  // -------- transformer layers --------
  for (int i = 0; i < TF_L; ++i) {
    const size_t wOff = (size_t)i * TF_H * TF_H * 2;
    const size_t bOff = (size_t)i * TF_H * 2;
    const size_t aOff = (size_t)i * TF_R * TF_H * 2;

    // adaLN modulation: [B, 6H]
    run_gemm(stream, mk(scond, TF_H, ada_w + (size_t)i * 6 * TF_H * TF_H * 2, TF_H,
                        ada_b + (size_t)i * 6 * TF_H * 2, modBuf, 6 * TF_H,
                        TF_B, 6 * TF_H, TF_H, F_BIAS, 1.f), 1);

    // merge LoRA into q/k/v weights
    merge_lora_kernel<<<(TF_H * TF_H + 255) / 256, 256, 0, stream>>>(
        wq_w + wOff, qb + aOff, qa + aOff, wqe, TF_LORA_SCALE);
    merge_lora_kernel<<<(TF_H * TF_H + 255) / 256, 256, 0, stream>>>(
        wk_w + wOff, kb + aOff, ka + aOff, wke, TF_LORA_SCALE);
    merge_lora_kernel<<<(TF_H * TF_H + 255) / 256, 256, 0, stream>>>(
        wv_w + wOff, vb + aOff, va + aOff, wve, TF_LORA_SCALE);

    // hm = modulate(cln(h), s1, sc1)
    cln_mod_kernel<<<TF_M, 256, 0, stream>>>(hBuf, hmBuf,
        modBuf + 0, modBuf + (size_t)TF_H * 2, 6 * TF_H * 2, TF_N);

    // q / k / v projections (LoRA-merged weights)
    run_gemm(stream, mk(hmBuf, TF_H, wqe, TF_H, wq_b + bOff, qBuf, TF_H,
                        TF_M, TF_H, TF_H, F_BIAS, 1.f), 1);
    run_gemm(stream, mk(hmBuf, TF_H, wke, TF_H, wk_b + bOff, kBuf, TF_H,
                        TF_M, TF_H, TF_H, F_BIAS, 1.f), 1);
    run_gemm(stream, mk(hmBuf, TF_H, wve, TF_H, wv_b + bOff, vBuf, TF_H,
                        TF_M, TF_H, TF_H, F_BIAS, 1.f), 1);

    // scores[z] = q[z] @ k[z] / sqrt(dh), z = b*NH + head
    {
      GemmP ps = mk(qBuf, TF_H, kBuf, TF_H, nullptr, scoreB, TF_N,
                    TF_N, TF_N, TF_DH, 0, 0.125f);
      ps.aDiv = TF_NH; ps.aSO = (long long)TF_N * TF_H; ps.aSI = TF_DH;
      ps.bDiv = TF_NH; ps.bSO = (long long)TF_N * TF_H; ps.bSI = TF_DH;
      ps.yDiv = 1;     ps.ySO = (long long)TF_N * TF_N; ps.ySI = 0;
      run_gemm(stream, ps, TF_B * TF_NH);
    }
    softmax2_kernel<<<TF_B * TF_NH * TF_N, 256, 0, stream>>>(scoreB);

    // out[z] = attn[z] @ v[z]   (B transposed: stored [key][dh])
    {
      GemmP pav = mk(scoreB, TF_N, vBuf, TF_H, nullptr, aoBuf, TF_H,
                     TF_N, TF_DH, TF_N, 0, 1.f);
      pav.aDiv = 1;     pav.aSO = (long long)TF_N * TF_N; pav.aSI = 0;
      pav.bTrans = 1;
      pav.bDiv = TF_NH; pav.bSO = (long long)TF_N * TF_H; pav.bSI = TF_DH;
      pav.yDiv = TF_NH; pav.ySO = (long long)TF_N * TF_H; pav.ySI = TF_DH;
      run_gemm(stream, pav, TF_B * TF_NH);
    }

    // h += g1 * (out @ wo + wo_b)
    {
      GemmP po = mk(aoBuf, TF_H, wo_w + wOff, TF_H, wo_b + bOff, hBuf, TF_H,
                    TF_M, TF_H, TF_H, F_BIAS | F_GATE, 1.f);
      po.gate = modBuf + (size_t)2 * TF_H * 2; po.gateStride = 6 * TF_H * 2;
      run_gemm(stream, po, 1);
    }

    // hm2 = modulate(cln(h), s2, sc2)
    cln_mod_kernel<<<TF_M, 256, 0, stream>>>(hBuf, hmBuf,
        modBuf + (size_t)3 * TF_H * 2, modBuf + (size_t)4 * TF_H * 2,
        6 * TF_H * 2, TF_N);

    // MLP with gated residual
    run_gemm(stream, mk(hmBuf, TF_H, m_w1 + (size_t)i * TF_MLP * TF_H * 2, TF_H,
                        m_b1 + (size_t)i * TF_MLP * 2, mlpBuf, TF_MLP,
                        TF_M, TF_MLP, TF_H, F_BIAS | F_SILU, 1.f), 1);
    {
      GemmP pm2 = mk(mlpBuf, TF_MLP, m_w2 + (size_t)i * TF_H * TF_MLP * 2, TF_MLP,
                     m_b2 + bOff, hBuf, TF_H,
                     TF_M, TF_H, TF_MLP, F_BIAS | F_GATE, 1.f);
      pm2.gate = modBuf + (size_t)5 * TF_H * 2; pm2.gateStride = 6 * TF_H * 2;
      run_gemm(stream, pm2, 1);
    }
  }

  // -------- final adaLN + projection --------
  run_gemm(stream, mk(scond, TF_H, fadaw, TF_H, fadab, fmodB, 2 * TF_H,
                      TF_B, 2 * TF_H, TF_H, F_BIAS, 1.f), 1);
  cln_mod_kernel<<<TF_M, 256, 0, stream>>>(hBuf, hmBuf,
      fmodB + 0, fmodB + (size_t)TF_H * 2, 2 * TF_H * 2, TF_N);
  run_gemm(stream, mk(hmBuf, TF_H, f_w, TF_H, f_b, (float*)d_out, TF_IN,
                      TF_M, TF_IN, TF_H, F_BIAS, 1.f), 1);
}
